// AttentionBlock_49349174231578
// MI455X (gfx1250) — compile-verified, matrix operand order checked by reading, not run on hardware
//
#include <hip/hip_runtime.h>
#include <hip/hip_bf16.h>

// ---------------------------------------------------------------------------
// AttentionBlock for MI455X (gfx1250): GroupNorm -> QKV -> SDPA -> proj + res
// All GEMMs run on v_wmma_f32_16x16x32_f16 (f16 in, f32 accumulate), with
// LDS-staged, double-buffered tiles so the matrix pipe (not L2) is the limit.
// B=16, C=512, H=W=32 -> N=1024 pixels, 3C=1536.
//
// GEMM blocking: 256 thr = 8 waves; block tile 256(M) x 128(N);
// wave tile 64x64 (acc[4][4], 16 wmma per 32-deep k-step).
// LDS: A 256x32 + B 128x32 f16, padded row stride 40 f16 (bank-conflict-free,
// 16B-aligned), double buffered = 60 KB (LDS/WGP = 320 KB).
// ---------------------------------------------------------------------------

typedef _Float16 v16h __attribute__((ext_vector_type(16)));
typedef _Float16 v8h  __attribute__((ext_vector_type(8)));
typedef float    v8f  __attribute__((ext_vector_type(8)));

#define BATCH 16
#define CCH   512
#define NPIX  1024
#define NGRP  8
#define CPG   (CCH / NGRP)     // 64
#define QKVO  (3 * CCH)        // 1536

#define LPAD  40               // padded LDS row stride in f16 (80 bytes)
#define BM    256              // block tile M
#define BN    128              // block tile N

// ---------------------------------------------------------------------------
// GroupNorm: one block per (batch, group). Pass 1: f32 sum/sumsq reduction.
// Pass 2: normalize + affine through a padded LDS transpose tile so both the
// global reads ([C,N] order) and writes ([N,C] order) are coalesced.
// Output: h16 in [B, N, C] (pixel-major, K-contiguous for the WMMA A operand).
// ---------------------------------------------------------------------------
__global__ void groupnorm_kernel(const float* __restrict__ x,
                                 const float* __restrict__ scale,
                                 const float* __restrict__ bias,
                                 _Float16* __restrict__ h16) {
    const int b = blockIdx.x / NGRP;
    const int g = blockIdx.x % NGRP;
    const int tid = threadIdx.x;
    const int nElem = CPG * NPIX;  // 65536
    const float* xg = x + ((size_t)b * CCH + (size_t)g * CPG) * NPIX;

    float s = 0.f, ss = 0.f;
    for (int i = tid; i < nElem; i += blockDim.x) {
        float v = xg[i];
        s += v;
        ss += v * v;
    }
    __shared__ float rs[256], rss[256];
    rs[tid] = s; rss[tid] = ss;
    __syncthreads();
    for (int off = 128; off > 0; off >>= 1) {
        if (tid < off) { rs[tid] += rs[tid + off]; rss[tid] += rss[tid + off]; }
        __syncthreads();
    }
    const float inv_n = 1.0f / (float)nElem;
    const float mean  = rs[0] * inv_n;
    const float var   = rss[0] * inv_n - mean * mean;
    const float rstd  = rsqrtf(var + 1e-5f);

    const int cw  = tid & 63;                    // invariant channel per thread
    const float sc = scale[g * CPG + cw] * rstd;
    const float bi = bias[g * CPG + cw];

    __shared__ float tile[64][65];
    for (int n0 = 0; n0 < NPIX; n0 += 64) {
        __syncthreads();
        for (int i = tid; i < 64 * 64; i += 256) {      // coalesced along n
            int c = i >> 6;
            int n = i & 63;
            tile[c][n] = xg[(size_t)c * NPIX + n0 + n];
        }
        __syncthreads();
        for (int i = tid; i < 64 * 64; i += 256) {      // coalesced along c
            int n = i >> 6;
            float v = (tile[cw][n] - mean) * sc + bi;
            h16[((size_t)b * NPIX + n0 + n) * CCH + g * CPG + cw] = (_Float16)v;
        }
    }
}

// ---------------------------------------------------------------------------
// Convert f32 weights to f16 (grid-stride)
// ---------------------------------------------------------------------------
__global__ void cvt_f32_f16_kernel(const float* __restrict__ w,
                                   _Float16* __restrict__ o, int n) {
    for (int i = blockIdx.x * blockDim.x + threadIdx.x; i < n;
         i += gridDim.x * blockDim.x)
        o[i] = (_Float16)w[i];
}

// ---------------------------------------------------------------------------
// Fragment load from an LDS tile row (CDNA5 16-bit A/B layout, wave32):
//   lane L: row = L & 15 (within 16-row subtile), klo = 8*(L>>4)
//   f16[0..7]  = row[klo .. klo+7], f16[8..15] = row[16+klo .. 16+klo+7]
// ---------------------------------------------------------------------------
__device__ __forceinline__ v16h frag_from_lds(const _Float16* rowp, int klo) {
    v8h lo = *(const v8h*)(rowp + klo);
    v8h hi = *(const v8h*)(rowp + 16 + klo);
    return __builtin_shufflevector(lo, hi, 0, 1, 2, 3, 4, 5, 6, 7,
                                           8, 9, 10, 11, 12, 13, 14, 15);
}

// ---------------------------------------------------------------------------
// LDS-staged, double-buffered 256x128 block GEMM core.
//   Ag: block A base (BM rows x K, leading dim lda, f16, row-major K)
//   Bg: block B base (BN rows x K, leading dim ldb) -- B operand stored [N][K]
// Each k-step: 256 threads stage 24 KB (A 16 KB + B 8 KB) -> LDS, one barrier,
// global loads for the next step overlap 16 wmma/wave from LDS.
// ---------------------------------------------------------------------------
__device__ __forceinline__ void staged_wmma_gemm(const _Float16* __restrict__ Ag,
                                                 int lda,
                                                 const _Float16* __restrict__ Bg,
                                                 int ldb, int K,
                                                 _Float16* __restrict__ lA,
                                                 _Float16* __restrict__ lB,
                                                 v8f acc[4][4]) {
    const int tid  = threadIdx.x;
    const int lane = tid & 31;
    const int wave = tid >> 5;
    const int idx  = lane & 15;
    const int klo  = (lane >> 4) << 3;       // 0 or 8
    const int mW   = (wave & 3) * 64;        // wave M offset in block tile
    const int nW   = (wave >> 2) * 64;       // wave N offset in block tile

    // Staging map: 16B chunks, coalesced (4 consecutive threads = one 64B row)
    int rowA[4], partA[4], rowB[2], partB[2];
#pragma unroll
    for (int i = 0; i < 4; ++i) {
        int c = tid + i * 256;
        rowA[i] = c >> 2; partA[i] = (c & 3) * 8;
    }
#pragma unroll
    for (int i = 0; i < 2; ++i) {
        int c = tid + i * 256;
        rowB[i] = c >> 2; partB[i] = (c & 3) * 8;
    }

    v8h ra[4], rb[2];
#pragma unroll
    for (int i = 0; i < 4; ++i)
        ra[i] = *(const v8h*)(Ag + (size_t)rowA[i] * lda + partA[i]);
#pragma unroll
    for (int i = 0; i < 2; ++i)
        rb[i] = *(const v8h*)(Bg + (size_t)rowB[i] * ldb + partB[i]);

    int cur = 0;
    for (int kk = 0; kk < K; kk += 32) {
        _Float16* sA = lA + cur * (BM * LPAD);
        _Float16* sB = lB + cur * (BN * LPAD);
#pragma unroll
        for (int i = 0; i < 4; ++i)
            *(v8h*)(sA + rowA[i] * LPAD + partA[i]) = ra[i];
#pragma unroll
        for (int i = 0; i < 2; ++i)
            *(v8h*)(sB + rowB[i] * LPAD + partB[i]) = rb[i];
        __syncthreads();

        if (kk + 32 < K) {  // prefetch next k-step while wmma runs
            const int kn = kk + 32;
#pragma unroll
            for (int i = 0; i < 4; ++i)
                ra[i] = *(const v8h*)(Ag + (size_t)rowA[i] * lda + kn + partA[i]);
#pragma unroll
            for (int i = 0; i < 2; ++i)
                rb[i] = *(const v8h*)(Bg + (size_t)rowB[i] * ldb + kn + partB[i]);
        }

        v16h af[4];
#pragma unroll
        for (int ia = 0; ia < 4; ++ia)
            af[ia] = frag_from_lds(sA + (mW + ia * 16 + idx) * LPAD, klo);
#pragma unroll
        for (int j = 0; j < 4; ++j) {
            v16h bf = frag_from_lds(sB + (nW + j * 16 + idx) * LPAD, klo);
            acc[0][j] = __builtin_amdgcn_wmma_f32_16x16x32_f16(false, af[0], false, bf,
                                                               (short)0, acc[0][j], false, false);
            acc[1][j] = __builtin_amdgcn_wmma_f32_16x16x32_f16(false, af[1], false, bf,
                                                               (short)0, acc[1][j], false, false);
            acc[2][j] = __builtin_amdgcn_wmma_f32_16x16x32_f16(false, af[2], false, bf,
                                                               (short)0, acc[2][j], false, false);
            acc[3][j] = __builtin_amdgcn_wmma_f32_16x16x32_f16(false, af[3], false, bf,
                                                               (short)0, acc[3][j], false, false);
        }
        cur ^= 1;
    }
}

// C/D layout: VGPR t, lane L: M = t + 8*(L>>4), N = L & 15.
// Output element (ia, j, t): row = ia*16 + 8*(lane>>4) + t, col = j*16 + (lane&15)
#define GEMM_PROLOGUE()                                           \
    const int lane = threadIdx.x & 31;                            \
    const int wave = threadIdx.x >> 5;                            \
    const int b = blockIdx.z;                                     \
    const int rowBase = blockIdx.y * BM + (wave & 3) * 64;        \
    const int colBase = blockIdx.x * BN + (wave >> 2) * 64;       \
    __shared__ _Float16 lA[2 * BM * LPAD];                        \
    __shared__ _Float16 lB[2 * BN * LPAD];                        \
    v8f acc[4][4] = {};                                           \
    (void)b;

#define EPI_ROW(ia, t) (rowBase + (ia) * 16 + (((lane) >> 4) << 3) + (t))
#define EPI_COL(j)     (colBase + (j) * 16 + ((lane) & 15))

// ---------------------------------------------------------------------------
// GEMM 1: QKV.  out[n, o] = sum_c h[n,c] * w[o,c] + b[o]
// Routes o<512 -> Q[b][n][c], 512..1023 -> K[b][n][c], 1024..1535 -> V^T[b][c][n]
// (V stored transposed so the attn@V B-operand is [Ncol][K] row-major.)
// ---------------------------------------------------------------------------
__global__ void __launch_bounds__(256)
gemm_qkv_kernel(const _Float16* __restrict__ h16,
                const _Float16* __restrict__ w16,
                const float* __restrict__ qkv_b,
                _Float16* __restrict__ q16,
                _Float16* __restrict__ k16,
                _Float16* __restrict__ vT16) {
    GEMM_PROLOGUE();
    staged_wmma_gemm(h16 + ((size_t)b * NPIX + blockIdx.y * BM) * CCH, CCH,
                     w16 + (size_t)blockIdx.x * BN * CCH, CCH, CCH, lA, lB, acc);

#pragma unroll
    for (int j = 0; j < 4; ++j) {
        const int o = EPI_COL(j);
        const float bias = qkv_b[o];
#pragma unroll
        for (int ia = 0; ia < 4; ++ia) {
#pragma unroll
            for (int t = 0; t < 8; ++t) {
                const int n = EPI_ROW(ia, t);
                const _Float16 v = (_Float16)(acc[ia][j][t] + bias);
                if (o < CCH) {
                    q16[((size_t)b * NPIX + n) * CCH + o] = v;
                } else if (o < 2 * CCH) {
                    k16[((size_t)b * NPIX + n) * CCH + (o - CCH)] = v;
                } else {
                    vT16[((size_t)b * CCH + (o - 2 * CCH)) * NPIX + n] = v;
                }
            }
        }
    }
}

// ---------------------------------------------------------------------------
// GEMM 2: scores.  s[n, m] = (1/sqrt(C)) * sum_c q[n,c] k[m,c]   (f32 out)
// ---------------------------------------------------------------------------
__global__ void __launch_bounds__(256)
gemm_scores_kernel(const _Float16* __restrict__ q16,
                   const _Float16* __restrict__ k16,
                   float* __restrict__ s32) {
    GEMM_PROLOGUE();
    const float scl = 0.044194173824159216f;  // 1/sqrt(512)
    staged_wmma_gemm(q16 + ((size_t)b * NPIX + blockIdx.y * BM) * CCH, CCH,
                     k16 + ((size_t)b * NPIX + blockIdx.x * BN) * CCH, CCH, CCH,
                     lA, lB, acc);

    float* sb = s32 + (size_t)b * NPIX * NPIX;
#pragma unroll
    for (int j = 0; j < 4; ++j) {
        const int m = EPI_COL(j);
#pragma unroll
        for (int ia = 0; ia < 4; ++ia) {
#pragma unroll
            for (int t = 0; t < 8; ++t)
                sb[(size_t)EPI_ROW(ia, t) * NPIX + m] = acc[ia][j][t] * scl;
        }
    }
}

// ---------------------------------------------------------------------------
// Softmax over each length-1024 row (f32 in, f16 probabilities out).
// ---------------------------------------------------------------------------
__global__ void softmax_kernel(const float* __restrict__ s32,
                               _Float16* __restrict__ p16) {
    const size_t row = blockIdx.x;
    const int tid = threadIdx.x;
    const float* sr = s32 + row * NPIX;

    float v[4];
    float m = -3.4e38f;
#pragma unroll
    for (int r = 0; r < 4; ++r) {
        v[r] = sr[tid + r * 256];
        m = fmaxf(m, v[r]);
    }
    __shared__ float red[256];
    red[tid] = m;
    __syncthreads();
    for (int off = 128; off > 0; off >>= 1) {
        if (tid < off) red[tid] = fmaxf(red[tid], red[tid + off]);
        __syncthreads();
    }
    const float rowmax = red[0];
    __syncthreads();

    float s = 0.f;
#pragma unroll
    for (int r = 0; r < 4; ++r) {
        v[r] = __expf(v[r] - rowmax);
        s += v[r];
    }
    red[tid] = s;
    __syncthreads();
    for (int off = 128; off > 0; off >>= 1) {
        if (tid < off) red[tid] += red[tid + off];
        __syncthreads();
    }
    const float inv = 1.0f / red[0];

    _Float16* pr = p16 + row * NPIX;
#pragma unroll
    for (int r = 0; r < 4; ++r)
        pr[tid + r * 256] = (_Float16)(v[r] * inv);
}

// ---------------------------------------------------------------------------
// GEMM 3: attn @ V.  o[n, c] = sum_m p[n,m] vT[c,m]   (f16 out for proj GEMM)
// ---------------------------------------------------------------------------
__global__ void __launch_bounds__(256)
gemm_av_kernel(const _Float16* __restrict__ p16,
               const _Float16* __restrict__ vT16,
               _Float16* __restrict__ o16) {
    GEMM_PROLOGUE();
    staged_wmma_gemm(p16 + ((size_t)b * NPIX + blockIdx.y * BM) * NPIX, NPIX,
                     vT16 + ((size_t)b * CCH + blockIdx.x * BN) * NPIX, NPIX, NPIX,
                     lA, lB, acc);

#pragma unroll
    for (int j = 0; j < 4; ++j) {
        const int c = EPI_COL(j);
#pragma unroll
        for (int ia = 0; ia < 4; ++ia) {
#pragma unroll
            for (int t = 0; t < 8; ++t)
                o16[((size_t)b * NPIX + EPI_ROW(ia, t)) * CCH + c] =
                    (_Float16)acc[ia][j][t];
        }
    }
}

// ---------------------------------------------------------------------------
// GEMM 4: proj + bias + residual.  out[b, o, n] = sum_c att[n,c] w[o,c]
//                                               + proj_b[o] + x[b, o, n]
// Final output layout [B, C, H, W] = [B, C, N] f32.
// ---------------------------------------------------------------------------
__global__ void __launch_bounds__(256)
gemm_proj_kernel(const _Float16* __restrict__ o16,
                 const _Float16* __restrict__ w16,
                 const float* __restrict__ proj_b,
                 const float* __restrict__ x,
                 float* __restrict__ out) {
    GEMM_PROLOGUE();
    staged_wmma_gemm(o16 + ((size_t)b * NPIX + blockIdx.y * BM) * CCH, CCH,
                     w16 + (size_t)blockIdx.x * BN * CCH, CCH, CCH, lA, lB, acc);

#pragma unroll
    for (int j = 0; j < 4; ++j) {
        const int o = EPI_COL(j);
        const float bias = proj_b[o];
#pragma unroll
        for (int ia = 0; ia < 4; ++ia) {
#pragma unroll
            for (int t = 0; t < 8; ++t) {
                const size_t idx = ((size_t)b * CCH + o) * NPIX + EPI_ROW(ia, t);
                out[idx] = acc[ia][j][t] + bias + x[idx];
            }
        }
    }
}

// ---------------------------------------------------------------------------
// Launch: workspace layout (bytes):
//   [0,16M)    h16   (later aliased by p16 [0,32M) after h,q dead)
//   [16M,32M)  q16
//   [32M,48M)  k16   (later aliased by o16)
//   [48M,64M)  vT16
//   [64M,128M) s32 (f32 scores)
//   [128M,129.5M) qkv_w f16 ; [129.5M,130M) proj_w f16
// Everything stays L2-resident (192 MB).
// ---------------------------------------------------------------------------
extern "C" void kernel_launch(void* const* d_in, const int* in_sizes, int n_in,
                              void* d_out, int out_size, void* d_ws, size_t ws_size,
                              hipStream_t stream) {
    const float* x        = (const float*)d_in[0];
    const float* gn_scale = (const float*)d_in[1];
    const float* gn_bias  = (const float*)d_in[2];
    const float* qkv_w    = (const float*)d_in[3];
    const float* qkv_b    = (const float*)d_in[4];
    const float* proj_w   = (const float*)d_in[5];
    const float* proj_b   = (const float*)d_in[6];
    float* out = (float*)d_out;

    char* ws = (char*)d_ws;
    const size_t MBs = 1ull << 20;
    _Float16* h16   = (_Float16*)(ws + 0);
    _Float16* q16   = (_Float16*)(ws + 16 * MBs);
    _Float16* k16   = (_Float16*)(ws + 32 * MBs);
    _Float16* vT16  = (_Float16*)(ws + 48 * MBs);
    float*    s32   = (float*)   (ws + 64 * MBs);
    _Float16* p16   = (_Float16*)(ws + 0);         // alias h16+q16 (dead)
    _Float16* o16   = (_Float16*)(ws + 32 * MBs);  // alias k16 (dead)
    _Float16* wq16  = (_Float16*)(ws + 128 * MBs);
    _Float16* wp16  = (_Float16*)(ws + 128 * MBs + (size_t)QKVO * CCH * 2);

    // Weight conversion
    cvt_f32_f16_kernel<<<256, 256, 0, stream>>>(qkv_w, wq16, QKVO * CCH);
    cvt_f32_f16_kernel<<<128, 256, 0, stream>>>(proj_w, wp16, CCH * CCH);

    // GroupNorm -> h16 [B, N, C]
    groupnorm_kernel<<<BATCH * NGRP, 256, 0, stream>>>(x, gn_scale, gn_bias, h16);

    // QKV GEMM: M=1024, Ncol=1536, K=512
    gemm_qkv_kernel<<<dim3(QKVO / BN, NPIX / BM, BATCH), 256, 0, stream>>>(
        h16, wq16, qkv_b, q16, k16, vT16);

    // Scores: M=1024, Ncol=1024, K=512
    gemm_scores_kernel<<<dim3(NPIX / BN, NPIX / BM, BATCH), 256, 0, stream>>>(
        q16, k16, s32);

    // Softmax over rows
    softmax_kernel<<<BATCH * NPIX, 256, 0, stream>>>(s32, p16);

    // attn @ V: M=1024, Ncol=512, K=1024
    gemm_av_kernel<<<dim3(CCH / BN, NPIX / BM, BATCH), 256, 0, stream>>>(
        p16, vT16, o16);

    // proj + bias + residual: M=1024, Ncol=512, K=512
    gemm_proj_kernel<<<dim3(CCH / BN, NPIX / BM, BATCH), 256, 0, stream>>>(
        o16, wp16, proj_b, x, out);
}
